// Incep_MHSA_85117661872663
// MI455X (gfx1250) — matrix operand order repaired and checked by reading, not compiled, not run on hardware
//
#include <hip/hip_runtime.h>
#include <hip/hip_bf16.h>

typedef __bf16 bf16_t;
typedef __attribute__((ext_vector_type(16))) __bf16 v16bf;
typedef __attribute__((ext_vector_type(8)))  __bf16 v8bf;
typedef __attribute__((ext_vector_type(8)))  float   v8f;
typedef __attribute__((ext_vector_type(4)))  unsigned int u32x4;
typedef __attribute__((ext_vector_type(8)))  int i32x8;
typedef __attribute__((ext_vector_type(4)))  int i32x4;

constexpr int BATCH = 8;
constexpr int CDIM  = 512;
constexpr int HGT   = 56;      // H == W
constexpr int NTOK  = 3136;    // 56*56
constexpr int NHEAD = 8;
constexpr int HDIM  = 64;
constexpr int MTOK  = 588;     // 3*196
constexpr int MPAD  = 608;     // multiple of 32
constexpr float QK_SCALE = 0.125f;   // 64^-0.5, exact power of two

__device__ inline v8f wmma_bf16(v16bf a, v16bf b, v8f c) {
  return __builtin_amdgcn_wmma_f32_16x16x32_bf16(false, a, false, b, (short)0, c,
                                                 false, false);
}

__device__ inline v16bf combine16(v8bf lo, v8bf hi) {
  v16bf r;
#pragma unroll
  for (int i = 0; i < 8; ++i) { r[i] = lo[i]; r[i + 8] = hi[i]; }
  return r;
}

// ---------------------------------------------------------------------------
// Kernel 0: one-time f32 -> bf16 conversion of Wq (512x512) and Wkv (1024x512)
// ---------------------------------------------------------------------------
__global__ __launch_bounds__(256) void prep_kernel(
    const float* __restrict__ Wq, const float* __restrict__ Wkv,
    bf16_t* __restrict__ Wqb, bf16_t* __restrict__ Wkvb) {
  const int i = blockIdx.x * 256 + threadIdx.x;
  if (i < CDIM * CDIM) Wqb[i] = (__bf16)Wq[i];
  if (i < 2 * CDIM * CDIM) Wkvb[i] = (__bf16)Wkv[i];
}

// ---------------------------------------------------------------------------
// Kernel 1: Q projection.  Q[b,n,o] = sum_c x[b,c,n]*Wq[o,c] + bq[o], * 0.125
// stored bf16 as [B, NTOK, 512].  x slab staged in LDS pre-swizzled into the
// WMMA A-operand layout (16 k-chunks x 32 lanes x 16 bf16).
// ---------------------------------------------------------------------------
__global__ __launch_bounds__(128) void qproj_kernel(
    const float* __restrict__ x, const bf16_t* __restrict__ Wqb,
    const float* __restrict__ bq, bf16_t* __restrict__ qg) {
  __shared__ bf16_t XshA[16 * 32 * 16];   // 16 KB, A-fragment layout
  const int b  = blockIdx.x / 196;
  const int nt = blockIdx.x % 196;
  const int n0 = nt * 16;
  const int tid = threadIdx.x;
  for (int idx = tid; idx < CDIM * 16; idx += 128) {
    const int c = idx >> 4, n = idx & 15;
    const float val = x[(size_t)(b * CDIM + c) * NTOK + n0 + n];
    const int kc = c >> 5, koff = c & 31;
    const int halfsel = koff >> 4;
    const int lane = n + (halfsel << 4);
    const int elem = koff & 15;
    XshA[(((kc << 5) + lane) << 4) + elem] = (__bf16)val;
  }
  __syncthreads();

  const int wave = tid >> 5, lane = tid & 31;
  const int ncol = lane & 15;
  const int ksel = (lane < 16) ? 0 : 16;
  const int halfo = (lane < 16) ? 0 : 8;

  for (int g = 0; g < 2; ++g) {
    const int ot0 = wave * 8 + g * 4;
    v8f acc[4] = {};
    for (int kc = 0; kc < 16; ++kc) {
      const int cb = kc * 32;
      const v16bf a = *(const v16bf*)(XshA + (((kc << 5) + lane) << 4));
#pragma unroll
      for (int t = 0; t < 4; ++t) {
        const int o = (ot0 + t) * 16 + ncol;
        const v16bf bb = *(const v16bf*)(Wqb + (size_t)o * CDIM + cb + ksel);
        acc[t] = wmma_bf16(a, bb, acc[t]);
      }
    }
#pragma unroll
    for (int t = 0; t < 4; ++t) {
      const int o = (ot0 + t) * 16 + ncol;
      const float bias = bq[o];
#pragma unroll
      for (int r = 0; r < 8; ++r) {
        const int nn = n0 + r + halfo;
        qg[((size_t)b * NTOK + nn) * CDIM + o] =
            (__bf16)((acc[t][r] + bias) * QK_SCALE);
      }
    }
  }
}

// ---------------------------------------------------------------------------
// Kernel 2: the three depthwise branches + concat -> l_pre[b,m,c] f32.
// ---------------------------------------------------------------------------
__global__ __launch_bounds__(256) void branches_kernel(
    const float* __restrict__ x,
    const float* __restrict__ w1a, const float* __restrict__ b1a,
    const float* __restrict__ w1b, const float* __restrict__ b1b,
    const float* __restrict__ w2,  const float* __restrict__ b2,
    const float* __restrict__ w3,  const float* __restrict__ b3,
    float* __restrict__ lpre) {
  __shared__ float Psh[16][16];   // pooled 14x14 with 1-pixel zero border
  const int b = blockIdx.x / CDIM;
  const int c = blockIdx.x % CDIM;
  const int t = threadIdx.x;
  const int ph = t / 14, pw = t % 14;
  const bool act = (t < 196);

  ((float*)Psh)[t] = 0.0f;
  __syncthreads();

  float l1 = 0.f, l2 = 0.f;
  if (act) {
    float xw[4][4];
    const float* xp = x + (size_t)(b * CDIM + c) * NTOK + (ph * 4) * HGT + pw * 4;
#pragma unroll
    for (int r = 0; r < 4; ++r)
#pragma unroll
      for (int s = 0; s < 4; ++s) xw[r][s] = xp[r * HGT + s];

    float a1 = b1b[c];
#pragma unroll
    for (int r = 0; r < 4; ++r) {
      float y = b1a[c];
#pragma unroll
      for (int s = 0; s < 4; ++s) y += w1a[c * 4 + s] * xw[r][s];
      y = fmaxf(y, 0.f);
      a1 += w1b[c * 4 + r] * y;
    }
    l1 = fmaxf(a1, 0.f);

    float a2 = b2[c];
#pragma unroll
    for (int r = 0; r < 4; ++r)
#pragma unroll
      for (int s = 0; s < 4; ++s) a2 += w2[c * 16 + r * 4 + s] * xw[r][s];
    l2 = fmaxf(a2, 0.f);

    float pool = 0.f;
#pragma unroll
    for (int r = 0; r < 4; ++r)
#pragma unroll
      for (int s = 0; s < 4; ++s) pool += xw[r][s];
    Psh[ph + 1][pw + 1] = pool * (1.0f / 16.0f);
  }
  __syncthreads();

  if (act) {
    float a3 = b3[c];
#pragma unroll
    for (int di = 0; di < 3; ++di)
#pragma unroll
      for (int dj = 0; dj < 3; ++dj)
        a3 += w3[c * 9 + di * 3 + dj] * Psh[ph + di][pw + dj];
    const float l3 = fmaxf(a3, 0.f);

    float* base = lpre + (size_t)b * MTOK * CDIM + c;
    base[(size_t)(0 * 196 + t) * CDIM] = l1;
    base[(size_t)(1 * 196 + t) * CDIM] = l2;
    base[(size_t)(2 * 196 + t) * CDIM] = l3;
  }
}

// ---------------------------------------------------------------------------
// Kernel 3: LayerNorm over C=512, output bf16.  Block = one (b,m) row.
// ---------------------------------------------------------------------------
__global__ __launch_bounds__(128) void ln_kernel(
    const float* __restrict__ lpre, const float* __restrict__ gamma,
    const float* __restrict__ beta, bf16_t* __restrict__ lno) {
  const int row = blockIdx.x;
  const float* p = lpre + (size_t)row * CDIM;
  const int t = threadIdx.x;
  float v[4], s = 0.f, s2 = 0.f;
#pragma unroll
  for (int i = 0; i < 4; ++i) {
    v[i] = p[t + i * 128];
    s += v[i];
    s2 += v[i] * v[i];
  }
  for (int off = 16; off >= 1; off >>= 1) {
    s  += __shfl_xor(s, off, 32);
    s2 += __shfl_xor(s2, off, 32);
  }
  __shared__ float shs[4], shs2[4];
  const int wave = t >> 5, lane = t & 31;
  if (lane == 0) { shs[wave] = s; shs2[wave] = s2; }
  __syncthreads();
  s  = shs[0] + shs[1] + shs[2] + shs[3];
  s2 = shs2[0] + shs2[1] + shs2[2] + shs2[3];
  const float mu = s * (1.0f / CDIM);
  const float var = s2 * (1.0f / CDIM) - mu * mu;
  const float rs = rsqrtf(var + 1e-5f);
  bf16_t* o = lno + (size_t)row * CDIM;
#pragma unroll
  for (int i = 0; i < 4; ++i) {
    const int cidx = t + i * 128;
    o[cidx] = (__bf16)((v[i] - mu) * rs * gamma[cidx] + beta[cidx]);
  }
}

// ---------------------------------------------------------------------------
// Kernel 4: KV projection GEMM (bf16 x bf16 -> f32 acc).
// o<512 -> K [B,h,588,64] bf16 ; o>=512 -> V transposed [B,h,64,588] bf16.
// ---------------------------------------------------------------------------
__global__ __launch_bounds__(128) void kv_kernel(
    const bf16_t* __restrict__ lno, const bf16_t* __restrict__ Wkvb,
    const float* __restrict__ bkv, bf16_t* __restrict__ Kg,
    bf16_t* __restrict__ Vtg) {
  const int b  = blockIdx.x / 37;
  const int mt = blockIdx.x % 37;
  const int tid = threadIdx.x, wave = tid >> 5, lane = tid & 31;
  const int mrow = lane & 15;
  const int kb   = (lane < 16) ? 0 : 16;
  const int ncol = lane & 15;
  const int ksel = (lane < 16) ? 0 : 16;
  const int halfo = (lane < 16) ? 0 : 8;
  const int m_a = mt * 16 + mrow;
  const bool avalid = (m_a < MTOK);
  const bf16_t* arow = lno + ((size_t)b * MTOK + (avalid ? m_a : 0)) * CDIM;

  v16bf afr[16];
#pragma unroll
  for (int kc = 0; kc < 16; ++kc) {
    const int cb = kc * 32;
    if (avalid) {
      const v8bf lo = *(const v8bf*)(arow + cb + kb);
      const v8bf hi = *(const v8bf*)(arow + cb + kb + 8);
      afr[kc] = combine16(lo, hi);
    } else {
#pragma unroll
      for (int i = 0; i < 16; ++i) afr[kc][i] = (__bf16)0.0f;
    }
  }

  for (int g = 0; g < 4; ++g) {
    const int ot0 = wave * 16 + g * 4;
    v8f acc[4] = {};
    for (int kc = 0; kc < 16; ++kc) {
      const int cb = kc * 32;
#pragma unroll
      for (int t2 = 0; t2 < 4; ++t2) {
        const int o = (ot0 + t2) * 16 + ncol;
        const bf16_t* wrow = Wkvb + (size_t)o * CDIM + cb + ksel;
        __builtin_prefetch(wrow + CDIM, 0, 1);
        const v16bf bb = *(const v16bf*)wrow;
        acc[t2] = wmma_bf16(afr[kc], bb, acc[t2]);
      }
    }
#pragma unroll
    for (int t2 = 0; t2 < 4; ++t2) {
      const int o = (ot0 + t2) * 16 + ncol;
      const float bias = bkv[o];
#pragma unroll
      for (int r = 0; r < 8; ++r) {
        const int m = mt * 16 + r + halfo;
        if (m < MTOK) {
          const float val = acc[t2][r] + bias;
          if (o < CDIM) {
            const int h = o >> 6, d = o & 63;
            Kg[(((size_t)(b * NHEAD + h)) * MTOK + m) * HDIM + d] = (__bf16)val;
          } else {
            const int oo = o - CDIM;
            const int h = oo >> 6, d = oo & 63;
            Vtg[(((size_t)(b * NHEAD + h)) * HDIM + d) * MTOK + m] = (__bf16)val;
          }
        }
      }
    }
  }
}

// ---------------------------------------------------------------------------
// Kernel 5: attention.  Block = (b,h, 64-query-row tile); 4 waves x 16 rows.
// K staged into LDS via the Tensor Data Mover (TDM) when available.
// ---------------------------------------------------------------------------
__global__ __launch_bounds__(128) void attn_kernel(
    const bf16_t* __restrict__ qg, const bf16_t* __restrict__ Kg,
    const bf16_t* __restrict__ Vtg, float* __restrict__ outp) {
  extern __shared__ char smem[];
  bf16_t* Ksh  = (bf16_t*)smem;                                // [MPAD][64]
  bf16_t* Vtsh = (bf16_t*)(smem + (size_t)MPAD * HDIM * 2);    // [64][MPAD]
  float*  Ssh  = (float*)(smem + (size_t)2 * MPAD * HDIM * 2); // [4][16][MPAD]

  const int bh = blockIdx.x / 49;
  const int nt = blockIdx.x % 49;
  const int b = bh >> 3, h = bh & 7;
  const int tid = threadIdx.x, wave = tid >> 5, lane = tid & 31;

  const bf16_t* Kglob = Kg + (size_t)bh * MTOK * HDIM;
#if __has_builtin(__builtin_amdgcn_tensor_load_to_lds) && defined(__AMDGCN__)
  // --- TDM: DMA the 588x64 bf16 K tile straight into LDS ---
  if (wave == 0) {
    const unsigned lds_off = (unsigned)(uintptr_t)(void*)Ksh;  // low 32b = LDS offset
    const unsigned long long ga = (unsigned long long)(uintptr_t)Kglob;
    u32x4 g0;
    g0[0] = 1u;                                   // count=1 (valid user D#)
    g0[1] = lds_off;                              // lds_addr
    g0[2] = (unsigned)(ga & 0xffffffffu);         // global_addr[31:0]
    g0[3] = (unsigned)((ga >> 32) & 0x01ffffffu) | (2u << 30);  // addr[56:32], type=2
    // group1: data_size=1(2B); tensor_dim0=64; tensor_dim1=588; tile 64x588;
    // dim0_stride=64 elements
    const unsigned long long w0 =
        (1ull << 16) | ((unsigned long long)(HDIM & 0xffff) << 48);
    const unsigned long long w1 =
        ((unsigned long long)HDIM >> 16) | ((unsigned long long)MTOK << 16) |
        ((unsigned long long)HDIM << 48);
    const unsigned long long w2 =
        (unsigned long long)MTOK | ((unsigned long long)HDIM << 32);
    const unsigned long long w3 = 0ull;
    i32x8 g1;
    g1[0] = (int)(w0 & 0xffffffffull); g1[1] = (int)(w0 >> 32);
    g1[2] = (int)(w1 & 0xffffffffull); g1[3] = (int)(w1 >> 32);
    g1[4] = (int)(w2 & 0xffffffffull); g1[5] = (int)(w2 >> 32);
    g1[6] = (int)(w3 & 0xffffffffull); g1[7] = (int)(w3 >> 32);
    const i32x4 gz4 = {0, 0, 0, 0};
    const i32x8 gz8 = {0, 0, 0, 0, 0, 0, 0, 0};
    __builtin_amdgcn_tensor_load_to_lds(g0, g1, gz4, gz4, gz8, 0);
    __builtin_amdgcn_s_wait_tensorcnt((short)0);
  }
  // zero the K pad rows j in [588,608)
  for (int idx = tid; idx < (MPAD - MTOK) * HDIM / 8; idx += 128) {
    const int j = MTOK + (idx >> 3);
    const int dc = (idx & 7) * 8;
    const uint4 z = {0u, 0u, 0u, 0u};
    *(uint4*)(Ksh + (size_t)j * HDIM + dc) = z;
  }
#else
  for (int idx = tid; idx < MPAD * HDIM / 8; idx += 128) {
    const int j = idx >> 3;
    const int dc = (idx & 7) * 8;
    uint4 val = {0u, 0u, 0u, 0u};
    if (j < MTOK) val = *(const uint4*)(Kglob + (size_t)j * HDIM + dc);
    *(uint4*)(Ksh + (size_t)j * HDIM + dc) = val;
  }
#endif
  // --- stage V^T rows (cols zero-padded), 8B chunks ---
  const bf16_t* Vglob = Vtg + (size_t)bh * HDIM * MTOK;
  for (int idx = tid; idx < HDIM * (MPAD / 4); idx += 128) {
    const int d = idx / (MPAD / 4);
    const int jc = (idx % (MPAD / 4)) * 4;
    uint2 val = {0u, 0u};
    if (jc + 3 < MTOK) {
      val = *(const uint2*)(Vglob + (size_t)d * MTOK + jc);
    } else {
      alignas(8) bf16_t tmp[4];
#pragma unroll
      for (int u = 0; u < 4; ++u)
        tmp[u] = (jc + u < MTOK) ? Vglob[(size_t)d * MTOK + jc + u] : (__bf16)0.0f;
      val = *(uint2*)tmp;
    }
    *(uint2*)(Vtsh + (size_t)d * MPAD + jc) = val;
  }
  __syncthreads();

  const int i0   = nt * 64 + wave * 16;
  const int mrow = lane & 15;
  const int kb   = (lane < 16) ? 0 : 16;
  const int ncol = lane & 15;
  const int ksel = (lane < 16) ? 0 : 16;
  const int halfo = (lane < 16) ? 0 : 8;

  const bf16_t* qrow = qg + ((size_t)b * NTOK + i0 + mrow) * CDIM + h * HDIM;
  v16bf aq[2];
#pragma unroll
  for (int c = 0; c < 2; ++c) {
    const v8bf lo = *(const v8bf*)(qrow + c * 32 + kb);
    const v8bf hi = *(const v8bf*)(qrow + c * 32 + kb + 8);
    aq[c] = combine16(lo, hi);
  }

  float* Sw = Ssh + (size_t)wave * 16 * MPAD;
  // --- scores S = (Q*scale) . K^T ---
  for (int jt = 0; jt < MPAD / 16; ++jt) {
    v8f s = {};
#pragma unroll
    for (int c = 0; c < 2; ++c) {
      const bf16_t* krow = Ksh + (size_t)(jt * 16 + ncol) * HDIM + c * 32 + ksel;
      const v16bf bb = *(const v16bf*)krow;
      s = wmma_bf16(aq[c], bb, s);
    }
#pragma unroll
    for (int r = 0; r < 8; ++r)
      Sw[(size_t)(r + halfo) * MPAD + jt * 16 + ncol] = s[r];
  }
  __syncthreads();

  // --- softmax row stats: lanes (r, r+16) split row r's 588 cols ---
  float rscale;
  {
    const int r = lane & 15;
    const int hsel = lane >> 4;
    float* srow = Sw + (size_t)r * MPAD;
    const int j0 = hsel * 294, j1 = j0 + 294;
    float mx = -1e30f;
    for (int j = j0; j < j1; ++j) mx = fmaxf(mx, srow[j]);
    mx = fmaxf(mx, __shfl_xor(mx, 16, 32));
    float sum = 0.f;
    for (int j = j0; j < j1; ++j) {
      const float pe = __expf(srow[j] - mx);
      srow[j] = pe;
      sum += pe;
    }
    sum += __shfl_xor(sum, 16, 32);
    if (hsel) {
      for (int j = MTOK; j < MPAD; ++j) srow[j] = 0.f;   // kill pad cols
    }
    rscale = 1.0f / sum;
  }
  __syncthreads();

  // --- O = P . V  (A = P rows from LDS f32 via float4 loads, B = V^T rows) ---
  v8f acc[4] = {};
  const float* prow = Sw + (size_t)mrow * MPAD;
  for (int jc = 0; jc < MPAD / 32; ++jc) {
    const float4 p0 = *(const float4*)(prow + jc * 32 + kb);
    const float4 p1 = *(const float4*)(prow + jc * 32 + kb + 4);
    const float4 p2 = *(const float4*)(prow + jc * 32 + kb + 8);
    const float4 p3 = *(const float4*)(prow + jc * 32 + kb + 12);
    v16bf a;
    a[0]  = (__bf16)(p0.x * rscale); a[1]  = (__bf16)(p0.y * rscale);
    a[2]  = (__bf16)(p0.z * rscale); a[3]  = (__bf16)(p0.w * rscale);
    a[4]  = (__bf16)(p1.x * rscale); a[5]  = (__bf16)(p1.y * rscale);
    a[6]  = (__bf16)(p1.z * rscale); a[7]  = (__bf16)(p1.w * rscale);
    a[8]  = (__bf16)(p2.x * rscale); a[9]  = (__bf16)(p2.y * rscale);
    a[10] = (__bf16)(p2.z * rscale); a[11] = (__bf16)(p2.w * rscale);
    a[12] = (__bf16)(p3.x * rscale); a[13] = (__bf16)(p3.y * rscale);
    a[14] = (__bf16)(p3.z * rscale); a[15] = (__bf16)(p3.w * rscale);
#pragma unroll
    for (int dt = 0; dt < 4; ++dt) {
      const bf16_t* vrow = Vtsh + (size_t)(dt * 16 + ncol) * MPAD + jc * 32 + ksel;
      const v16bf bb = *(const v16bf*)vrow;
      acc[dt] = wmma_bf16(a, bb, acc[dt]);
    }
  }

#pragma unroll
  for (int dt = 0; dt < 4; ++dt)
#pragma unroll
    for (int r = 0; r < 8; ++r) {
      const int i = i0 + r + halfo;
      outp[((size_t)b * NTOK + i) * CDIM + h * HDIM + dt * 16 + ncol] = acc[dt][r];
    }
}

// ---------------------------------------------------------------------------
extern "C" void kernel_launch(void* const* d_in, const int* in_sizes, int n_in,
                              void* d_out, int out_size, void* d_ws,
                              size_t ws_size, hipStream_t stream) {
  (void)in_sizes; (void)n_in; (void)out_size; (void)ws_size;
  const float* x    = (const float*)d_in[0];
  const float* Wq   = (const float*)d_in[1];
  const float* bq   = (const float*)d_in[2];
  const float* Wkv  = (const float*)d_in[3];
  const float* bkv  = (const float*)d_in[4];
  const float* w1a  = (const float*)d_in[5];
  const float* b1a  = (const float*)d_in[6];
  const float* w1b  = (const float*)d_in[7];
  const float* b1b  = (const float*)d_in[8];
  const float* w2   = (const float*)d_in[9];
  const float* b2   = (const float*)d_in[10];
  const float* w3   = (const float*)d_in[11];
  const float* b3   = (const float*)d_in[12];
  const float* gam  = (const float*)d_in[13];
  const float* bet  = (const float*)d_in[14];

  char* ws = (char*)d_ws;
  const size_t off_q    = 0;                                            // bf16 [8,3136,512]
  const size_t off_lpre = off_q    + (size_t)BATCH * NTOK * CDIM * 2;   // f32 [8,588,512]
  const size_t off_ln   = off_lpre + (size_t)BATCH * MTOK * CDIM * 4;   // bf16
  const size_t off_k    = off_ln   + (size_t)BATCH * MTOK * CDIM * 2;   // bf16 [8,8,588,64]
  const size_t off_vt   = off_k    + (size_t)BATCH * NHEAD * MTOK * HDIM * 2;
  const size_t off_wqb  = off_vt   + (size_t)BATCH * NHEAD * MTOK * HDIM * 2;
  const size_t off_wkvb = off_wqb  + (size_t)CDIM * CDIM * 2;

  bf16_t* qg   = (bf16_t*)(ws + off_q);
  float*  lpre = (float*)(ws + off_lpre);
  bf16_t* lno  = (bf16_t*)(ws + off_ln);
  bf16_t* Kgp  = (bf16_t*)(ws + off_k);
  bf16_t* Vtgp = (bf16_t*)(ws + off_vt);
  bf16_t* Wqb  = (bf16_t*)(ws + off_wqb);
  bf16_t* Wkvb = (bf16_t*)(ws + off_wkvb);
  float*  outp = (float*)d_out;

  prep_kernel<<<dim3((2 * CDIM * CDIM + 255) / 256), 256, 0, stream>>>(
      Wq, Wkv, Wqb, Wkvb);
  qproj_kernel<<<dim3(BATCH * 196), 128, 0, stream>>>(x, Wqb, bq, qg);
  branches_kernel<<<dim3(BATCH * CDIM), 256, 0, stream>>>(
      x, w1a, b1a, w1b, b1b, w2, b2, w3, b3, lpre);
  ln_kernel<<<dim3(BATCH * MTOK), 128, 0, stream>>>(lpre, gam, bet, lno);
  kv_kernel<<<dim3(BATCH * 37), 128, 0, stream>>>(lno, Wkvb, bkv, Kgp, Vtgp);

  const size_t smem = (size_t)2 * MPAD * HDIM * 2 + (size_t)4 * 16 * MPAD * 4;
  attn_kernel<<<dim3(BATCH * NHEAD * 49), 128, smem, stream>>>(qg, Kgp, Vtgp,
                                                               outp);
}
